// GCNInformer_7627861917723
// MI455X (gfx1250) — matrix operand order repaired and checked by reading, not compiled, not run on hardware
//
#include <hip/hip_runtime.h>
#include <hip/hip_bf16.h>

// ---------------------------------------------------------------------------
// Shapes (fixed by the reference)
// ---------------------------------------------------------------------------
#define T_   512
#define N_   512
#define EMB_ 64
#define GC_  32
#define E_   4096

typedef __attribute__((ext_vector_type(16))) _Float16 v16h;
typedef __attribute__((ext_vector_type(8)))  _Float16 v8h;
typedef __attribute__((ext_vector_type(8)))  float    v8f;
typedef __attribute__((ext_vector_type(4)))  float    v4f;

// ---------------------------------------------------------------------------
// WMMA helpers. CDNA5 16x16x32 f16 WMMA, wave32.
// VGPR layout assumptions (ISA 7.12.2):
//  A (16x32 f16): lane L<16 -> row L,   k in {0..7, 16..23}
//                 lane L>=16-> row L-16, k in {8..15,24..31}
//  B (32x16 f16): lane L<16 -> col L,   k in {0..7, 16..23}
//                 lane L>=16-> col L-16, k in {8..15,24..31}  (B stored as [n][k])
//  C (16x16 f32): vgpr r, lane L -> row r + 8*(L>=16), col L&15
// ---------------------------------------------------------------------------
__device__ __forceinline__ v16h pack16(v8h lo, v8h hi) {
  v16h r;
#pragma unroll
  for (int i = 0; i < 8; ++i) { r[i] = lo[i]; r[i + 8] = hi[i]; }
  return r;
}

// A fragment: mat is row-major [m][k] with leading dim ld (in halves).
__device__ __forceinline__ v16h load_frag_A(const _Float16* mat, size_t ld,
                                            int row0, int k0) {
  int lane = threadIdx.x & 31;
  const _Float16* p = mat + (size_t)(row0 + (lane & 15)) * ld + k0 +
                      ((lane & 16) ? 8 : 0);
  return pack16(*(const v8h*)p, *(const v8h*)(p + 16));
}

// B fragment: mat is "n-major" [n][k] with leading dim ld (i.e. B transposed).
__device__ __forceinline__ v16h load_frag_B(const _Float16* mat, size_t ld,
                                            int n0, int k0) {
  int lane = threadIdx.x & 31;
  const _Float16* p = mat + (size_t)(n0 + (lane & 15)) * ld + k0 +
                      ((lane & 16) ? 8 : 0);
  return pack16(*(const v8h*)p, *(const v8h*)(p + 16));
}

__device__ __forceinline__ v8f wmma_f16(v16h a, v16h b, v8f c) {
  return __builtin_amdgcn_wmma_f32_16x16x32_f16(false, a, false, b, (short)0, c,
                                                false, false);
}

// ---------------------------------------------------------------------------
// Kernel A: Q = xWq + bq, K = xWk + bk (f16 outputs), plus f16 transposed x.
// Grid: T*N/64 blocks, 128 threads (4 waves, 16 rows per wave).
// ---------------------------------------------------------------------------
__global__ __launch_bounds__(128) void k_proj(
    const float* __restrict__ x, const float* __restrict__ Wq,
    const float* __restrict__ bq, const float* __restrict__ Wk,
    const float* __restrict__ bk, _Float16* __restrict__ Qh,
    _Float16* __restrict__ Kh, _Float16* __restrict__ xth) {
  __shared__ __align__(16) _Float16 sWqT[64 * 72];  // Wq^T : [n][k]
  __shared__ __align__(16) _Float16 sWkT[64 * 72];
  __shared__ __align__(16) _Float16 sX[64 * 72];    // x tile: [m][k]
  int tid = threadIdx.x;
  for (int i = tid; i < 64 * 64; i += 128) {
    int r = i >> 6, c = i & 63;
    sWqT[c * 72 + r] = (_Float16)Wq[i];
    sWkT[c * 72 + r] = (_Float16)Wk[i];
  }
  size_t R = (size_t)blockIdx.x * 64;  // base row in flattened [T*N, EMB]
  for (int i = tid; i < 64 * 64; i += 128) {
    int r = i >> 6, c = i & 63;
    sX[r * 72 + c] = (_Float16)x[(R + r) * 64 + c];
  }
  __syncthreads();

  int wave = tid >> 5, lane = tid & 31;
  int rowbase = wave * 16;
  v16h a0 = load_frag_A(sX, 72, rowbase, 0);
  v16h a1 = load_frag_A(sX, 72, rowbase, 32);
  int n = lane & 15;
  int mh = (lane & 16) ? 8 : 0;
  for (int nt = 0; nt < 4; ++nt) {
    v8f accq = {};
    accq = wmma_f16(a0, load_frag_B(sWqT, 72, nt * 16, 0), accq);
    accq = wmma_f16(a1, load_frag_B(sWqT, 72, nt * 16, 32), accq);
    v8f acck = {};
    acck = wmma_f16(a0, load_frag_B(sWkT, 72, nt * 16, 0), acck);
    acck = wmma_f16(a1, load_frag_B(sWkT, 72, nt * 16, 32), acck);
    float biasq = bq[nt * 16 + n];
    float biask = bk[nt * 16 + n];
#pragma unroll
    for (int r = 0; r < 8; ++r) {
      size_t grow = R + rowbase + r + mh;
      Qh[grow * 64 + nt * 16 + n] = (_Float16)(accq[r] + biasq);
      Kh[grow * 64 + nt * 16 + n] = (_Float16)(acck[r] + biask);
    }
  }
  __syncthreads();
  // xth[t][e][u] = f16(x[t][u][e]) — "n-major" B panel for attn @ x.
  int t = (int)(R >> 9);
  int u0 = (int)(R & 511);
  for (int e0 = 0; e0 < 64; e0 += 2) {
    int e = e0 + (tid >> 6);
    int i = tid & 63;
    xth[((size_t)t * 64 + e) * 512 + u0 + i] = sX[i * 72 + e];
  }
}

// gWth[t][c][e] = f16(gcn_W[t][e][c]) — B panel for cx @ gcn_W.
__global__ __launch_bounds__(256) void k_convw(const float* __restrict__ gW,
                                               _Float16* __restrict__ gWth) {
  int i = blockIdx.x * 256 + threadIdx.x;
  if (i >= T_ * 64 * 32) return;
  int t = i / (64 * 32);
  int rem = i - t * (64 * 32);
  int e = rem >> 5, c = rem & 31;
  gWth[((size_t)t * 32 + c) * 64 + e] = (_Float16)gW[i];
}

// ---------------------------------------------------------------------------
// Kernel B (hot): scores = QK^T/8 -> softmax -> stream attn (d_out, NT) ->
//                 cx = attn @ x -> h = cx @ gcn_W.  Grid (T, N/16), 128 thr.
// Qh/Kh/xth panels (~96MB) are L2-resident (192MB L2) -> direct global frags.
// The 512MB attn stream is written NON-TEMPORAL so it can't evict the panels.
// ---------------------------------------------------------------------------
__global__ __launch_bounds__(128) void k_attn(
    const _Float16* __restrict__ Qh, const _Float16* __restrict__ Kh,
    const _Float16* __restrict__ xth, const _Float16* __restrict__ gWth,
    float* __restrict__ attn_out, float* __restrict__ h_ws) {
  __shared__ __align__(16) float sS[16 * 516];     // score/softmax tile
  __shared__ __align__(16) _Float16 sA[16 * 528];  // attn tile as f16 A panel
  __shared__ __align__(16) _Float16 sCx[16 * 72];  // cx tile as f16 A panel
  __shared__ float sRed[16 * 8];
  __shared__ float sRow[16];

  int t = blockIdx.x;
  int s0 = blockIdx.y * 16;
  int tid = threadIdx.x, wave = tid >> 5, lane = tid & 31;
  int n = lane & 15, mh = (lane & 16) ? 8 : 0;

  const _Float16* Qt = Qh + (size_t)t * N_ * EMB_;
  const _Float16* Kt = Kh + (size_t)t * N_ * EMB_;
  v16h a0 = load_frag_A(Qt, 64, s0, 0);
  v16h a1 = load_frag_A(Qt, 64, s0, 32);
  // scores: 32 column tiles split across 4 waves
  for (int nt = wave; nt < 32; nt += 4) {
    v8f acc = {};
    acc = wmma_f16(a0, load_frag_B(Kt, 64, nt * 16, 0), acc);
    acc = wmma_f16(a1, load_frag_B(Kt, 64, nt * 16, 32), acc);
#pragma unroll
    for (int r = 0; r < 8; ++r)
      sS[(r + mh) * 516 + nt * 16 + n] = acc[r] * 0.125f;  // 1/sqrt(64)
  }
  __syncthreads();

  // softmax over 512 columns: 8 threads per row, 64 elems each
  int row = tid & 15, chunk = tid >> 4;
  {
    float mx = -3.4e38f;
    for (int j = 0; j < 64; ++j) mx = fmaxf(mx, sS[row * 516 + chunk * 64 + j]);
    sRed[row * 8 + chunk] = mx;
  }
  __syncthreads();
  if (chunk == 0) {
    float mx = sRed[row * 8];
    for (int j = 1; j < 8; ++j) mx = fmaxf(mx, sRed[row * 8 + j]);
    sRow[row] = mx;
  }
  __syncthreads();
  {
    float mx = sRow[row], sum = 0.f;
    for (int j = 0; j < 64; ++j) {
      float e = __expf(sS[row * 516 + chunk * 64 + j] - mx);
      sS[row * 516 + chunk * 64 + j] = e;
      sum += e;
    }
    sRed[row * 8 + chunk] = sum;
  }
  __syncthreads();
  if (chunk == 0) {
    float s = 0.f;
    for (int j = 0; j < 8; ++j) s += sRed[row * 8 + j];
    sRow[row] = 1.0f / s;
  }
  __syncthreads();
  // normalize in LDS + build f16 A panel
  {
    float inv = sRow[row];
    for (int j = 0; j < 64; ++j) {
      int u = chunk * 64 + j;
      float v = sS[row * 516 + u] * inv;
      sS[row * 516 + u] = v;
      sA[row * 528 + u] = (_Float16)v;
    }
  }
  __syncthreads();
  // Stream attn tile to global, non-temporal, alignment-fixed vectors.
  // attn starts at d_out+1 so every row base is ≡1 (mod 4 floats): store
  // cols 0..2 and 511 as scalars, cols 3..510 as 16B-aligned float4 NT.
  {
    int r2 = tid >> 3;  // 16 rows, 8 threads each
    int s2 = tid & 7;
    float* aout = attn_out + ((size_t)t * N_ + s0 + r2) * N_;
    const float* srow = &sS[r2 * 516];
    if (s2 == 0) {
      __builtin_nontemporal_store(srow[0], aout + 0);
      __builtin_nontemporal_store(srow[1], aout + 1);
      __builtin_nontemporal_store(srow[2], aout + 2);
      __builtin_nontemporal_store(srow[511], aout + 511);
    }
    for (int q = s2; q < 127; q += 8) {
      int c0 = 3 + 4 * q;
      v4f v = {srow[c0], srow[c0 + 1], srow[c0 + 2], srow[c0 + 3]};
      __builtin_nontemporal_store(v, (v4f*)(aout + c0));
    }
  }
  __syncthreads();

  // cx = attn @ x : wave w owns EMB columns [16w, 16w+16)
  {
    v8f acc = {};
    const _Float16* xt = xth + (size_t)t * 64 * 512;
    for (int kk = 0; kk < 16; ++kk) {
      v16h aa = load_frag_A(sA, 528, 0, kk * 32);
      v16h bb = load_frag_B(xt, 512, wave * 16, kk * 32);
      acc = wmma_f16(aa, bb, acc);
    }
#pragma unroll
    for (int r = 0; r < 8; ++r)
      sCx[(r + mh) * 72 + wave * 16 + n] = (_Float16)acc[r];
  }
  __syncthreads();

  // h = cx @ gcn_W[t] : waves 0,1 own GC column tiles
  if (wave < 2) {
    const _Float16* gwt = gWth + (size_t)t * 32 * 64;
    v16h c0 = load_frag_A(sCx, 72, 0, 0);
    v16h c1 = load_frag_A(sCx, 72, 0, 32);
    v8f acc = {};
    acc = wmma_f16(c0, load_frag_B(gwt, 64, wave * 16, 0), acc);
    acc = wmma_f16(c1, load_frag_B(gwt, 64, wave * 16, 32), acc);
#pragma unroll
    for (int r = 0; r < 8; ++r)
      h_ws[((size_t)t * N_ + s0 + r + mh) * GC_ + wave * 16 + n] = acc[r];
  }
}

// ---------------------------------------------------------------------------
// Kernel C: GCN symmetric-normalized aggregation in LDS (64KB agg fits in
// the 320KB WGP LDS; ds_add_f32 atomics, no global atomics). Grid: T blocks.
// ---------------------------------------------------------------------------
__global__ __launch_bounds__(256) void k_gcn(const int* __restrict__ eidx,
                                             const float* __restrict__ h_ws,
                                             const float* __restrict__ gcn_b,
                                             float* __restrict__ seq) {
  __shared__ float agg[N_ * GC_];  // 64KB
  __shared__ float deg[N_];
  int t = blockIdx.x, tid = threadIdx.x;
  const int* ei = eidx + (size_t)t * 2 * E_;
  for (int i = tid; i < N_; i += 256) deg[i] = 1.0f;  // self loop
  __syncthreads();
  for (int e = tid; e < E_; e += 256) {
    int dst = ei[E_ + e] & (N_ - 1);
    atomicAdd(&deg[dst], 1.0f);
  }
  __syncthreads();
  const float* hbase = h_ws + (size_t)t * N_ * GC_;
  for (int nn = tid; nn < N_; nn += 256) {  // self-loop term: norm = 1/deg
    float inv = 1.0f / deg[nn];
    const v4f* hr = (const v4f*)(hbase + (size_t)nn * GC_);
#pragma unroll
    for (int c4 = 0; c4 < GC_ / 4; ++c4) {
      v4f h4 = hr[c4];
#pragma unroll
      for (int k = 0; k < 4; ++k) agg[nn * GC_ + c4 * 4 + k] = h4[k] * inv;
    }
  }
  __syncthreads();
  for (int e = tid; e < E_; e += 256) {
    int src = ei[e] & (N_ - 1);
    int dst = ei[E_ + e] & (N_ - 1);
    float norm = rsqrtf(deg[src]) * rsqrtf(deg[dst]);
    const v4f* hr = (const v4f*)(hbase + (size_t)src * GC_);
#pragma unroll
    for (int c4 = 0; c4 < GC_ / 4; ++c4) {
      v4f h4 = hr[c4];
#pragma unroll
      for (int k = 0; k < 4; ++k)
        atomicAdd(&agg[dst * GC_ + c4 * 4 + k], h4[k] * norm);
    }
  }
  __syncthreads();
  const float* bt = gcn_b + t * GC_;
  float* st = seq + (size_t)t * N_ * GC_;
  for (int i4 = tid; i4 < (N_ * GC_) / 4; i4 += 256) {
    int i = i4 * 4;
    v4f b4 = *(const v4f*)(bt + (i & 31));
    v4f v;
#pragma unroll
    for (int k = 0; k < 4; ++k) v[k] = fmaxf(agg[i + k] + b4[k], 0.0f);
    *(v4f*)(st + i) = v;  // relu(agg + b) -> seq
  }
}

// ---------------------------------------------------------------------------
// Kernel D: gate preactivations G[t][j] = seq[t] . w_ih[j] + b_ih[j]+b_hh[j]
// ---------------------------------------------------------------------------
__global__ __launch_bounds__(256) void k_gates(const float* __restrict__ seq,
                                               const float* __restrict__ w_ih,
                                               const float* __restrict__ b_ih,
                                               const float* __restrict__ b_hh,
                                               float* __restrict__ G) {
  __shared__ float red[4 * 256];
  int t = blockIdx.x, tid = threadIdx.x;
  float p0 = 0, p1 = 0, p2 = 0, p3 = 0;
  const float* s = seq + (size_t)t * 16384;
  for (int d = tid * 4; d < 16384; d += 256 * 4) {
    v4f v = *(const v4f*)(s + d);
    v4f w0 = *(const v4f*)(w_ih + d);
    v4f w1 = *(const v4f*)(w_ih + 16384 + d);
    v4f w2 = *(const v4f*)(w_ih + 2 * 16384 + d);
    v4f w3 = *(const v4f*)(w_ih + 3 * 16384 + d);
#pragma unroll
    for (int k = 0; k < 4; ++k) {
      p0 += v[k] * w0[k];
      p1 += v[k] * w1[k];
      p2 += v[k] * w2[k];
      p3 += v[k] * w3[k];
    }
  }
  red[tid] = p0; red[256 + tid] = p1; red[512 + tid] = p2; red[768 + tid] = p3;
  __syncthreads();
  for (int off = 128; off > 0; off >>= 1) {
    if (tid < off) {
      red[tid] += red[tid + off];
      red[256 + tid] += red[256 + tid + off];
      red[512 + tid] += red[512 + tid + off];
      red[768 + tid] += red[768 + tid + off];
    }
    __syncthreads();
  }
  if (tid == 0) {
    G[t * 4 + 0] = red[0] + b_ih[0] + b_hh[0];
    G[t * 4 + 1] = red[256] + b_ih[1] + b_hh[1];
    G[t * 4 + 2] = red[512] + b_ih[2] + b_hh[2];
    G[t * 4 + 3] = red[768] + b_ih[3] + b_hh[3];
  }
}

// ---------------------------------------------------------------------------
// Kernel E: sequential LSTM recurrence (H=1) + final FCs. One thread.
// ---------------------------------------------------------------------------
__global__ void k_lstm(const float* __restrict__ G,
                       const float* __restrict__ w_hh,
                       const float* __restrict__ fc1_w,
                       const float* __restrict__ fc1_b,
                       const float* __restrict__ fc2_w,
                       const float* __restrict__ fc2_b,
                       float* __restrict__ out) {
  if (threadIdx.x != 0 || blockIdx.x != 0) return;
  float h = 0.f, c = 0.f;
  float wi = w_hh[0], wf = w_hh[1], wg = w_hh[2], wo = w_hh[3];
  for (int t = 0; t < T_; ++t) {
    float gi = G[t * 4 + 0] + wi * h;
    float gf = G[t * 4 + 1] + wf * h;
    float gg = G[t * 4 + 2] + wg * h;
    float go = G[t * 4 + 3] + wo * h;
    float si = 1.f / (1.f + __expf(-gi));
    float sf = 1.f / (1.f + __expf(-gf));
    float so = 1.f / (1.f + __expf(-go));
    c = sf * c + si * tanhf(gg);
    h = so * tanhf(c);
  }
  float inf = h * fc1_w[0] + fc1_b[0];
  out[0] = inf * fc2_w[0] + fc2_b[0];
}

// ---------------------------------------------------------------------------
extern "C" void kernel_launch(void* const* d_in, const int* in_sizes, int n_in,
                              void* d_out, int out_size, void* d_ws,
                              size_t ws_size, hipStream_t stream) {
  const float* x    = (const float*)d_in[0];
  const int*   eidx = (const int*)d_in[1];
  const float* Wq   = (const float*)d_in[2];
  const float* bq   = (const float*)d_in[3];
  const float* Wk   = (const float*)d_in[4];
  const float* bk   = (const float*)d_in[5];
  const float* gW   = (const float*)d_in[6];
  const float* gb   = (const float*)d_in[7];
  const float* w_ih = (const float*)d_in[8];
  const float* w_hh = (const float*)d_in[9];
  const float* b_ih = (const float*)d_in[10];
  const float* b_hh = (const float*)d_in[11];
  const float* fc1w = (const float*)d_in[12];
  const float* fc1b = (const float*)d_in[13];
  const float* fc2w = (const float*)d_in[14];
  const float* fc2b = (const float*)d_in[15];
  float* out = (float*)d_out;
  float* attn_out = out + 1;  // tuple: (scalar out, attn[T,N,N])

  char* ws = (char*)d_ws;
  size_t off = 0;
  auto alloc = [&](size_t bytes) {
    void* p = ws + off;
    off = (off + bytes + 255) & ~(size_t)255;
    return p;
  };
  _Float16* Qh   = (_Float16*)alloc((size_t)T_ * N_ * EMB_ * 2);  // 32MB
  _Float16* Kh   = (_Float16*)alloc((size_t)T_ * N_ * EMB_ * 2);  // 32MB
  _Float16* xth  = (_Float16*)alloc((size_t)T_ * EMB_ * N_ * 2);  // 32MB
  _Float16* gWth = (_Float16*)alloc((size_t)T_ * GC_ * EMB_ * 2); // 2MB
  float*    h_ws = (float*)alloc((size_t)T_ * N_ * GC_ * 4);      // 32MB
  float*    seq  = (float*)alloc((size_t)T_ * N_ * GC_ * 4);      // 32MB
  float*    G    = (float*)alloc((size_t)T_ * 4 * 4);
  (void)in_sizes; (void)n_in; (void)out_size; (void)ws_size;

  k_proj<<<dim3((T_ * N_) / 64), 128, 0, stream>>>(x, Wq, bq, Wk, bk, Qh, Kh,
                                                   xth);
  k_convw<<<dim3((T_ * 64 * 32 + 255) / 256), 256, 0, stream>>>(gW, gWth);
  k_attn<<<dim3(T_, N_ / 16), 128, 0, stream>>>(Qh, Kh, xth, gWth, attn_out,
                                                h_ws);
  k_gcn<<<dim3(T_), 256, 0, stream>>>(eidx, h_ws, gb, seq);
  k_gates<<<dim3(T_), 256, 0, stream>>>(seq, w_ih, b_ih, b_hh, G);
  k_lstm<<<dim3(1), 32, 0, stream>>>(G, w_hh, fc1w, fc1b, fc2w, fc2b, out);
}
